// my_RNN_63591285784578
// MI455X (gfx1250) — matrix operand order repaired and checked by reading, not compiled, not run on hardware
//
#include <hip/hip_runtime.h>
#include <hip/hip_bf16.h>
#include <stdint.h>

#define IN_F  1024
#define HID   1024
#define OUT_F 1024
#define BATCH 64
#define SEQ   512

// LDS row stride (bf16 elems) for GEMM staging tiles: 80B rows, 16B-aligned chunks
#define LSTR 40
// LDS row stride for the resident hidden state in the scan kernel:
// 1032 elems = 2064B -> 16B aligned, 4-dword bank skew between rows
#define HSTR 1032

typedef __attribute__((ext_vector_type(16))) __bf16 bf16x16;
typedef __attribute__((ext_vector_type(8)))  __bf16 bf16x8;
typedef __attribute__((ext_vector_type(8)))  float  f32x8;
typedef __attribute__((ext_vector_type(4)))  float  f32x4;

union FragU { bf16x16 v; bf16x8 h[2]; };

// Fast tanh: prefer the CDNA5 hardware V_TANH_F32 transcendental.
__device__ __forceinline__ float fast_tanh(float x) {
#if __has_builtin(__builtin_amdgcn_tanhf)
  return __builtin_amdgcn_tanhf(x);
#else
  // tanh(x) = 1 - 2/(exp(2x)+1), via v_exp_f32 + v_rcp_f32
  float e = __builtin_amdgcn_exp2f(x * 2.8853900817779268f /* 2*log2(e) */);
  return 1.0f - 2.0f * __builtin_amdgcn_rcpf(e + 1.0f);
#endif
}

// A-matrix fragment (16x32 bf16, row-major, given row stride).
// Lane l: m = l%16, half = l/16.
//   elements 0..15 = K (half*16 .. half*16+15)
__device__ __forceinline__ bf16x16 frag_a(const __bf16* base, int stride, int lane) {
  const int m = lane & 15, half = lane >> 4;
  const __bf16* p = base + m * stride;
  FragU f;
  f.h[0] = *reinterpret_cast<const bf16x8*>(p + half * 16);
  f.h[1] = *reinterpret_cast<const bf16x8*>(p + half * 16 + 8);
  return f.v;
}

// B-matrix fragment (32x16 bf16). Weight rows are the N dimension (W[n][k]).
// Lane l: n = l%16, half = l/16, elements 0..15 = K (half*16 .. half*16+15).
__device__ __forceinline__ bf16x16 frag_b(const __bf16* base, int stride, int lane) {
  const int n = lane & 15, half = lane >> 4;
  const __bf16* p = base + n * stride + half * 16;
  FragU f;
  f.h[0] = *reinterpret_cast<const bf16x8*>(p);
  f.h[1] = *reinterpret_cast<const bf16x8*>(p + 8);
  return f.v;
}

__device__ __forceinline__ f32x8 wmma_bf16(bf16x16 a, bf16x16 b, f32x8 c) {
  return __builtin_amdgcn_wmma_f32_16x16x32_bf16(
      /*neg_a=*/false, a, /*neg_b=*/false, b,
      /*c_mod=*/(short)0, c, /*reuse_a=*/false, /*reuse_b=*/false);
}

// ---------------------------------------------------------------------------
// fp32 -> bf16 strided conversion (weight slices, initial hidden state)
// dst[r*cols + c] = src[r*src_stride + src_off + c]
// ---------------------------------------------------------------------------
__global__ __launch_bounds__(256) void k_cvt(const float* __restrict__ src,
                                             __bf16* __restrict__ dst,
                                             int rows, int cols,
                                             int src_stride, int src_off) {
  int i = blockIdx.x * blockDim.x + threadIdx.x;
  if (i >= rows * cols) return;
  int r = i / cols, c = i - r * cols;
  dst[i] = (__bf16)src[(size_t)r * src_stride + src_off + c];
}

// ---------------------------------------------------------------------------
// Pack W_h (fp32, row n = output feature, col k) into B-fragment order:
//   dst[((nt*32 + kc)*2 + chunk)*256 + lane*8 + e]
//     = W_h[nt*16 + lane%16][kc*32 + (lane/16)*16 + chunk*8 + e]
// Each wave B-fragment load then becomes two fully-coalesced 512B bursts.
// ---------------------------------------------------------------------------
__global__ __launch_bounds__(256) void k_packW(const float* __restrict__ src,
                                               __bf16* __restrict__ dst,
                                               int src_stride, int src_off) {
  int i = blockIdx.x * blockDim.x + threadIdx.x;
  if (i >= HID * HID) return;
  int e    = i & 7;
  int lane = (i >> 3) & 31;
  int c    = (i >> 8) & 1;
  int kc   = (i >> 9) & 31;
  int nt   = i >> 14;
  int n = nt * 16 + (lane & 15);
  int k = kc * 32 + (lane >> 4) * 16 + c * 8 + e;
  dst[i] = (__bf16)src[(size_t)n * src_stride + src_off + k];
}

// ---------------------------------------------------------------------------
// K1: x_proj = x . Wx^T + b_ih, stored in C-FRAGMENT-PACKED fp32 layout:
//   xp[(( (s*4 + mt) * 64 + nt ) * 32 + lane) * 8 + j]
// where mt = b/16 for row s*64+b, nt = h/16, and (lane, j) is the 16x16 f32 C
// layout (lane = (m%16>=8)*16 + n%16, j = m%8). Each K1 lane already holds
// exactly this fragment, so the store is one contiguous f32x8 per lane/tile,
// and the scan kernel's accumulator seed is a contiguous 32B load per tile.
// Block = 256 thr (8 waves), tile 128M x 128N, K step 32.
// ---------------------------------------------------------------------------
__global__ __launch_bounds__(256) void k_xproj(const float*  __restrict__ x,
                                               const __bf16* __restrict__ Wx,
                                               const float*  __restrict__ bih,
                                               float*        __restrict__ xp) {
  __shared__ __bf16 As[128 * LSTR];
  __shared__ __bf16 Bs[128 * LSTR];
  const int tid  = threadIdx.x;
  const int lane = tid & 31;
  const int wave = tid >> 5;
  const int wm   = (wave >> 1) * 32;  // 4 waves along M
  const int wn   = (wave & 1) * 64;   // 2 waves along N
  const int bm0  = blockIdx.x * 128;
  const int bn0  = blockIdx.y * 128;

  const f32x8 zero = {};
  f32x8 acc[2][4];
#pragma unroll
  for (int i = 0; i < 2; ++i)
#pragma unroll
    for (int j = 0; j < 4; ++j) acc[i][j] = zero;

  for (int k0 = 0; k0 < IN_F; k0 += 32) {
    // Stage A: 128x32 fp32 -> bf16 into LDS (convert on the fly)
#pragma unroll
    for (int pass = 0; pass < 4; ++pass) {
      int r  = pass * 32 + (tid >> 3);
      int c4 = (tid & 7) * 4;
      int m  = bm0 + r;
      int s  = m >> 6, b = m & 63;  // m = s*64 + b
      f32x4 v = *reinterpret_cast<const f32x4*>(
          x + ((size_t)b * SEQ + s) * IN_F + k0 + c4);
      __bf16* d = As + r * LSTR + c4;
      d[0] = (__bf16)v[0]; d[1] = (__bf16)v[1];
      d[2] = (__bf16)v[2]; d[3] = (__bf16)v[3];
    }
    // Stage B: 128x32 bf16 weights
    {
      int r = tid >> 1, hc = (tid & 1) * 16;
      const __bf16* s_ = Wx + (size_t)(bn0 + r) * IN_F + k0 + hc;
      *reinterpret_cast<bf16x8*>(Bs + r * LSTR + hc)     =
          *reinterpret_cast<const bf16x8*>(s_);
      *reinterpret_cast<bf16x8*>(Bs + r * LSTR + hc + 8) =
          *reinterpret_cast<const bf16x8*>(s_ + 8);
    }
    __syncthreads();

    bf16x16 af[2], bfm[4];
#pragma unroll
    for (int sm = 0; sm < 2; ++sm) af[sm]  = frag_a(As + (wm + sm * 16) * LSTR, LSTR, lane);
#pragma unroll
    for (int sn = 0; sn < 4; ++sn) bfm[sn] = frag_b(Bs + (wn + sn * 16) * LSTR, LSTR, lane);
#pragma unroll
    for (int sm = 0; sm < 2; ++sm)
#pragma unroll
      for (int sn = 0; sn < 4; ++sn)
        acc[sm][sn] = wmma_bf16(af[sm], bfm[sn], acc[sm][sn]);
    __syncthreads();
  }

  // Fragment-packed epilogue store (bias folded in)
#pragma unroll
  for (int sm = 0; sm < 2; ++sm) {
    int gm0  = bm0 + wm + sm * 16;     // row-tile base (multiple of 16)
    int s4mt = gm0 >> 4;               // = s*4 + batch-tile
#pragma unroll
    for (int sn = 0; sn < 4; ++sn) {
      int gn0 = bn0 + wn + sn * 16;
      int nt  = gn0 >> 4;
      float bias = bih[gn0 + (lane & 15)];
      f32x8 v;
#pragma unroll
      for (int j = 0; j < 8; ++j) v[j] = acc[sm][sn][j] + bias;
      *reinterpret_cast<f32x8*>(
          xp + ((((size_t)s4mt * 64 + nt) * 32) + lane) * 8) = v;
    }
  }
}

// ---------------------------------------------------------------------------
// K2: persistent scan kernel. ONE launch for all 512 steps.
// Recurrence is independent per batch row, so grid = 4 workgroups, each owning
// 16 batch rows (one WMMA M-tile) and its full 16x1024 hidden state in LDS.
// No inter-workgroup sync ever; one workgroup barrier pair per step.
// Per wave: N slice of 128 columns (8 WMMA tiles), K loop of 32.
// B fragments come straight from L2 via the pre-packed W_h (no LDS staging);
// accumulator seeds come from the fragment-packed x_proj (contiguous 32B).
// ---------------------------------------------------------------------------
__global__ __launch_bounds__(256) void k_scan(const __bf16* __restrict__ h0,
                                              const __bf16* __restrict__ Whp,
                                              const float*  __restrict__ xp,
                                              __bf16*       __restrict__ Hbuf) {
  __shared__ __bf16 hs[16 * HSTR];  // resident hidden state, 33KB
  const int tid   = threadIdx.x;
  const int lane  = tid & 31;
  const int wave  = tid >> 5;
  const int mt    = blockIdx.x;          // batch tile (16 rows) owned by this WG
  const int b0    = mt * 16;
  const int n0    = wave * 128;          // this wave's N slice
  const int m0    = (lane >> 4) * 8;     // C-fragment row base for this lane
  const int nlane = lane & 15;           // C-fragment column for this lane

  // Load initial hidden state (16 x 1024 bf16) into LDS
  for (int idx = tid; idx < 16 * 128; idx += 256) {
    int r = idx >> 7, c8 = (idx & 127) * 8;
    *reinterpret_cast<bf16x8*>(&hs[r * HSTR + c8]) =
        *reinterpret_cast<const bf16x8*>(h0 + (size_t)(b0 + r) * HID + c8);
  }
  __syncthreads();

  for (int s = 0; s < SEQ; ++s) {
    __bf16* hout = Hbuf + ((size_t)s * BATCH + b0) * HID;
    const float* xps = xp + (((size_t)(s * 4 + mt) * 64) * 32) * 8;

    // Seed accumulators with fragment-packed x_proj (bias folded in K1)
    f32x8 acc[8];
#pragma unroll
    for (int sn = 0; sn < 8; ++sn) {
      int nt = (n0 >> 4) + sn;
      acc[sn] = *reinterpret_cast<const f32x8*>(xps + ((size_t)nt * 32 + lane) * 8);
    }

    // acc += h_prev . Wh^T  (A from LDS, B from packed global/L2)
    for (int kc = 0; kc < 32; ++kc) {
      bf16x16 af = frag_a(hs + kc * 32, HSTR, lane);
#pragma unroll
      for (int sn = 0; sn < 8; ++sn) {
        int nt = (n0 >> 4) + sn;
        const __bf16* p = Whp + (((size_t)nt * 32 + kc) * 2) * 256 + lane * 8;
        FragU f;
        f.h[0] = *reinterpret_cast<const bf16x8*>(p);
        f.h[1] = *reinterpret_cast<const bf16x8*>(p + 256);
        acc[sn] = wmma_bf16(af, f.v, acc[sn]);
      }
    }

    __syncthreads();  // all waves done READING hs before in-place update

#pragma unroll
    for (int sn = 0; sn < 8; ++sn) {
      int n = n0 + sn * 16 + nlane;
#pragma unroll
      for (int j = 0; j < 8; ++j) {
        __bf16 hb = (__bf16)fast_tanh(acc[sn][j]);     // V_TANH_F32
        hs[(m0 + j) * HSTR + n] = hb;                  // next step's A
        hout[(size_t)(m0 + j) * HID + n] = hb;         // for output GEMM
      }
    }

    __syncthreads();  // hs fully updated before next step reads it
  }
}

// ---------------------------------------------------------------------------
// K3: out[b][s][o] = H[s*64+b][:] . Who[o][:] + b_ho[o]   (fp32 out)
// Same 128x128 tiling as K1; A is already bf16 row-major.
// ---------------------------------------------------------------------------
__global__ __launch_bounds__(256) void k_out(const __bf16* __restrict__ H,
                                             const __bf16* __restrict__ Who,
                                             const float*  __restrict__ bho,
                                             float*        __restrict__ out) {
  __shared__ __bf16 As[128 * LSTR];
  __shared__ __bf16 Bs[128 * LSTR];
  const int tid  = threadIdx.x;
  const int lane = tid & 31;
  const int wave = tid >> 5;
  const int wm   = (wave >> 1) * 32;
  const int wn   = (wave & 1) * 64;
  const int bm0  = blockIdx.x * 128;
  const int bn0  = blockIdx.y * 128;

  const f32x8 zero = {};
  f32x8 acc[2][4];
#pragma unroll
  for (int i = 0; i < 2; ++i)
#pragma unroll
    for (int j = 0; j < 4; ++j) acc[i][j] = zero;

  for (int k0 = 0; k0 < HID; k0 += 32) {
    int r = tid >> 1, hc = (tid & 1) * 16;
    {
      const __bf16* s_ = H + (size_t)(bm0 + r) * HID + k0 + hc;
      *reinterpret_cast<bf16x8*>(As + r * LSTR + hc)     =
          *reinterpret_cast<const bf16x8*>(s_);
      *reinterpret_cast<bf16x8*>(As + r * LSTR + hc + 8) =
          *reinterpret_cast<const bf16x8*>(s_ + 8);
    }
    {
      const __bf16* s_ = Who + (size_t)(bn0 + r) * HID + k0 + hc;
      *reinterpret_cast<bf16x8*>(Bs + r * LSTR + hc)     =
          *reinterpret_cast<const bf16x8*>(s_);
      *reinterpret_cast<bf16x8*>(Bs + r * LSTR + hc + 8) =
          *reinterpret_cast<const bf16x8*>(s_ + 8);
    }
    __syncthreads();

    bf16x16 af[2], bfm[4];
#pragma unroll
    for (int sm = 0; sm < 2; ++sm) af[sm]  = frag_a(As + (wm + sm * 16) * LSTR, LSTR, lane);
#pragma unroll
    for (int sn = 0; sn < 4; ++sn) bfm[sn] = frag_b(Bs + (wn + sn * 16) * LSTR, LSTR, lane);
#pragma unroll
    for (int sm = 0; sm < 2; ++sm)
#pragma unroll
      for (int sn = 0; sn < 4; ++sn)
        acc[sm][sn] = wmma_bf16(af[sm], bfm[sn], acc[sm][sn]);
    __syncthreads();
  }

#pragma unroll
  for (int sm = 0; sm < 2; ++sm) {
    int gm0 = bm0 + wm + sm * 16 + (lane >> 4) * 8;
#pragma unroll
    for (int sn = 0; sn < 4; ++sn) {
      int gn = bn0 + wn + sn * 16 + (lane & 15);
      float bias = bho[gn];
#pragma unroll
      for (int j = 0; j < 8; ++j) {
        int gm = gm0 + j;
        int s = gm >> 6, b = gm & 63;  // H row = s*64 + b ; out is [b][s][o]
        out[((size_t)b * SEQ + s) * OUT_F + gn] = acc[sm][sn][j] + bias;
      }
    }
  }
}

// ---------------------------------------------------------------------------
// Workspace layout (bytes), total ~200 MB:
//   [0,   2M)   Wx   bf16 (1024x1024)
//   [2M,  4M)   Whp  bf16 (1024x1024, B-fragment packed)
//   [4M,  6M)   Who  bf16 (1024x1024)
//   [6M,  8M)   h0   bf16 (64x1024, from hidden_first)
//   [8M,  72M)  Hbuf bf16 (512 x 64 x 1024)  -- all hidden states, row-major
//   [72M, 200M) xp   fp32 (512 x 64 x 1024)  -- x_proj, C-fragment packed
// ---------------------------------------------------------------------------
extern "C" void kernel_launch(void* const* d_in, const int* in_sizes, int n_in,
                              void* d_out, int out_size, void* d_ws, size_t ws_size,
                              hipStream_t stream) {
  (void)in_sizes; (void)n_in; (void)out_size; (void)ws_size;
  const float* x      = (const float*)d_in[0];
  const float* hfirst = (const float*)d_in[1];
  const float* Wih    = (const float*)d_in[2];
  const float* bih    = (const float*)d_in[3];
  const float* Whof   = (const float*)d_in[4];
  const float* bho    = (const float*)d_in[5];
  float* out = (float*)d_out;

  char* ws = (char*)d_ws;
  __bf16* Wx    = (__bf16*)(ws);
  __bf16* Whp   = (__bf16*)(ws + (size_t)2 * 1024 * 1024);
  __bf16* Who   = (__bf16*)(ws + (size_t)4 * 1024 * 1024);
  __bf16* h0    = (__bf16*)(ws + (size_t)6 * 1024 * 1024);
  __bf16* Hbuf  = (__bf16*)(ws + (size_t)8 * 1024 * 1024);
  float*  xp    = (float*) (ws + (size_t)8 * 1024 * 1024 +
                            (size_t)SEQ * BATCH * HID * sizeof(__bf16));

  const int CT = 256;
  // Weight conversions / packing (W_ih is [HID][IN_F+HID]: split into Wx, Wh)
  k_cvt<<<(HID * IN_F + CT - 1) / CT, CT, 0, stream>>>(Wih, Wx, HID, IN_F, IN_F + HID, 0);
  k_packW<<<(HID * HID + CT - 1) / CT, CT, 0, stream>>>(Wih, Whp, IN_F + HID, IN_F);
  k_cvt<<<(OUT_F * HID + CT - 1) / CT, CT, 0, stream>>>(Whof, Who, OUT_F, HID, HID, 0);
  k_cvt<<<(BATCH * HID + CT - 1) / CT, CT, 0, stream>>>(hfirst, h0, BATCH, HID, HID, 0);

  // K1: big input-projection GEMM (parallel over all timesteps)
  dim3 g1(SEQ * BATCH / 128, HID / 128);
  k_xproj<<<g1, 256, 0, stream>>>(x, Wx, bih, xp);

  // K2: full 512-step recurrence in ONE persistent launch
  // (4 workgroups x 16 batch rows; no inter-WG dependencies)
  k_scan<<<BATCH / 16, 256, 0, stream>>>(h0, Whp, xp, Hbuf);

  // K3: big output GEMM over all stored hidden states
  dim3 g3(SEQ * BATCH / 128, OUT_F / 128);
  k_out<<<g3, 256, 0, stream>>>(Hbuf, Who, bho, out);
}
